// GPT_46531675685500
// MI455X (gfx1250) — matrix operand order repaired and checked by reading, not compiled, not run on hardware
//
#include <hip/hip_runtime.h>

// ---------------- model constants ----------------
#define L_  6
#define E_  512
#define H_  8
#define V_  32000
#define B_  2
#define T_  2048
#define DH_ 64
#define E3_ (3*E_)
#define E4_ (4*E_)

typedef __attribute__((ext_vector_type(16))) __bf16 v16bf;
typedef __attribute__((ext_vector_type(8)))  float  v8f;

union V16U { uint4 u[2]; v16bf v; };
union U4H  { uint4 u; __bf16 h[8]; };

__device__ __forceinline__ v16bf ld16(const __bf16* p0, const __bf16* p1) {
    V16U t;
    t.u[0] = *(const uint4*)p0;
    t.u[1] = *(const uint4*)p1;
    return t.v;
}

// ---- CDNA5 async global->LDS copy (16B per lane), tracked by ASYNCcnt ----
__device__ __forceinline__ void async_copy16(__bf16* lds_dst, const __bf16* gsrc) {
    unsigned lds = (unsigned)(size_t)lds_dst;      // low 32 bits of shared ptr = LDS offset
    asm volatile("global_load_async_to_lds_b128 %0, %1, off"
                 :: "v"(lds), "v"(gsrc) : "memory");
}
__device__ __forceinline__ void async_wait0() {
    asm volatile("s_wait_asynccnt 0x0" ::: "memory");
}

// ---------------- f32 -> bf16 weight conversion ----------------
__global__ __launch_bounds__(256)
void convert_kernel(const float* __restrict__ in, __bf16* __restrict__ out, int n)
{
    int i = blockIdx.x * 256 + threadIdx.x;
    if (i < n) out[i] = (__bf16)in[i];
}

// ---------------- embedding: x = wte[idx] + wpe ----------------
__global__ __launch_bounds__(256)
void embed_kernel(const int* __restrict__ idx, const float* __restrict__ wte,
                  const float* __restrict__ wpe, float* __restrict__ X)
{
    size_t i = (size_t)blockIdx.x * 256 + threadIdx.x;   // over B*T*E
    int c  = (int)(i % E_);
    size_t bt = i / E_;
    int t  = (int)(bt % T_);
    X[i] = wte[(size_t)idx[bt] * E_ + c] + wpe[(size_t)t * E_ + c];
}

// ------- LayerNorm: one wave per row of E=512, emits bf16 for WMMA -------
__global__ __launch_bounds__(256)
void ln_kernel(const float* __restrict__ X, const float* __restrict__ g,
               const float* __restrict__ bb, __bf16* __restrict__ Out, int rows)
{
    int gw   = (int)((blockIdx.x * 256 + threadIdx.x) >> 5);
    int lane = threadIdx.x & 31;
    if (gw >= rows) return;
    const float* xr = X + (size_t)gw * E_;
    float v[16];
    float s = 0.0f;
#pragma unroll
    for (int j = 0; j < 16; j++) { v[j] = xr[lane + 32*j]; s += v[j]; }
#pragma unroll
    for (int off = 16; off >= 1; off >>= 1) s += __shfl_xor(s, off, 32);
    float mu = s * (1.0f / E_);
    float sq = 0.0f;
#pragma unroll
    for (int j = 0; j < 16; j++) { float d = v[j] - mu; sq += d * d; }
#pragma unroll
    for (int off = 16; off >= 1; off >>= 1) sq += __shfl_xor(sq, off, 32);
    float rstd = rsqrtf(sq * (1.0f / E_) + 1e-5f);
    __bf16* orow = Out + (size_t)gw * E_;
#pragma unroll
    for (int j = 0; j < 16; j++) {
        int c = lane + 32*j;
        orow[c] = (__bf16)((v[j] - mu) * rstd * g[c] + bb[c]);
    }
}

// ---- LayerNorm of only the last token of each batch (for lm_head) ----
__global__ void lnf_last_kernel(const float* __restrict__ X, const float* __restrict__ g,
                                const float* __restrict__ bb, float* __restrict__ Out)
{
    int b = threadIdx.x >> 5, lane = threadIdx.x & 31;
    const float* xr = X + ((size_t)b * T_ + (T_ - 1)) * E_;
    float v[16];
    float s = 0.0f;
#pragma unroll
    for (int j = 0; j < 16; j++) { v[j] = xr[lane + 32*j]; s += v[j]; }
#pragma unroll
    for (int off = 16; off >= 1; off >>= 1) s += __shfl_xor(s, off, 32);
    float mu = s * (1.0f / E_);
    float sq = 0.0f;
#pragma unroll
    for (int j = 0; j < 16; j++) { float d = v[j] - mu; sq += d * d; }
#pragma unroll
    for (int off = 16; off >= 1; off >>= 1) sq += __shfl_xor(sq, off, 32);
    float rstd = rsqrtf(sq * (1.0f / E_) + 1e-5f);
#pragma unroll
    for (int j = 0; j < 16; j++) {
        int c = lane + 32*j;
        Out[(size_t)b * E_ + c] = (v[j] - mu) * rstd * g[c] + bb[c];
    }
}

// ---------------- bf16 WMMA GEMM (double-buffered async staging) ----------------
// A:[M,K] bf16 row-major, Bw:[K,N] bf16 row-major, bias f32.
// C = op(A@B + bias) (+resid); OUTBF16 selects bf16 vs f32 output.
// Block tile 128x128, K-step 32, 256 threads = 8 waves (2x4).
template<bool GELU, bool RESID, bool OUTBF16>
__global__ __launch_bounds__(256)
void gemm_kernel(const __bf16* __restrict__ A, const __bf16* __restrict__ Bw,
                 const float* __restrict__ bias, const float* __restrict__ resid,
                 void* __restrict__ Cout, int M, int N, int K)
{
    const int LDA = 48, LDB = 48;                 // bf16 strides, 96B (16B aligned)
    __shared__ __bf16 As[2][128 * 48];
    __shared__ __bf16 Bs[2][128 * 48];            // transposed: Bs[n][k]

    int tid  = threadIdx.x;
    int lane = tid & 31;
    int wave = tid >> 5;
    int wm = wave >> 2;                           // 0..1  (64 rows each)
    int wn = wave & 3;                            // 0..3  (32 cols each)
    int m0 = blockIdx.y * 128;
    int n0 = blockIdx.x * 128;

    auto stageA = [&](int buf, int k0) {
#pragma unroll
        for (int j = 0; j < 2; j++) {
            int u = tid + 256 * j;
            int row = u >> 2, ch = u & 3;
            async_copy16(&As[buf][row * LDA + ch * 8],
                         A + (size_t)(m0 + row) * K + k0 + ch * 8);
        }
    };
    auto stageB = [&](int buf, int k0) {
#pragma unroll
        for (int j = 0; j < 2; j++) {
            int u = tid + 256 * j;
            int kk = u >> 4, nch = u & 15;
            U4H w;
            w.u = *(const uint4*)(Bw + (size_t)(k0 + kk) * N + n0 + nch * 8);
#pragma unroll
            for (int i = 0; i < 8; i++)
                Bs[buf][(nch * 8 + i) * LDB + kk] = w.h[i];
        }
    };

    v8f acc[4][2];
#pragma unroll
    for (int i = 0; i < 4; i++)
#pragma unroll
        for (int j = 0; j < 2; j++) acc[i][j] = {};

    int mlane = lane & 15;
    int ah = (lane >> 4) * 8;                     // A k-offset 0/8
    int bh = (lane >> 4) * 8;                     // B k-offset 0/8

    stageA(0, 0);
    stageB(0, 0);
    int cur = 0;
    for (int k0 = 0; k0 < K; k0 += 32) {
        async_wait0();                            // buffer `cur` fully landed in LDS
        __syncthreads();
        if (k0 + 32 < K) {                        // prefetch next tile while computing
            stageA(cur ^ 1, k0 + 32);
            stageB(cur ^ 1, k0 + 32);
        }

        const __bf16* Asb = As[cur];
        const __bf16* Bsb = Bs[cur];
        v16bf aop[4], bop[2];
#pragma unroll
        for (int im = 0; im < 4; im++) {
            int r = wm * 64 + im * 16 + mlane;
            aop[im] = ld16(&Asb[r * LDA + ah], &Asb[r * LDA + ah + 16]);
        }
#pragma unroll
        for (int in_ = 0; in_ < 2; in_++) {
            int cn = wn * 32 + in_ * 16 + mlane;
            bop[in_] = ld16(&Bsb[cn * LDB + bh], &Bsb[cn * LDB + bh + 16]);
        }
#pragma unroll
        for (int im = 0; im < 4; im++)
#pragma unroll
            for (int in_ = 0; in_ < 2; in_++)
                acc[im][in_] = __builtin_amdgcn_wmma_f32_16x16x32_bf16(
                    false, aop[im], false, bop[in_], (short)0, acc[im][in_], false, false);
        cur ^= 1;
    }

    // epilogue: C layout -> lanes 0-15: N=lane,M=e ; lanes 16-31: N=lane-16,M=8+e
    int mh = (lane >> 4) * 8;
#pragma unroll
    for (int im = 0; im < 4; im++) {
#pragma unroll
        for (int in_ = 0; in_ < 2; in_++) {
            int ng = n0 + wn * 32 + in_ * 16 + mlane;
            float bia = bias[ng];
#pragma unroll
            for (int e = 0; e < 8; e++) {
                int mg = m0 + wm * 64 + im * 16 + mh + e;
                float vv = acc[im][in_][e] + bia;
                if (GELU) vv = 0.5f * vv * (1.0f + erff(vv * 0.70710678118654752f));
                if (RESID) vv += resid[(size_t)mg * N + ng];
                if (OUTBF16) ((__bf16*)Cout)[(size_t)mg * N + ng] = (__bf16)vv;
                else         ((float*) Cout)[(size_t)mg * N + ng] = vv;
            }
        }
    }
}

// ------- split qkv bf16 [B,T,3E] -> q,k,v bf16 [B,H,T,DH] (pure reshape) -------
__global__ __launch_bounds__(256)
void split_kernel(const __bf16* __restrict__ qkv, __bf16* __restrict__ q,
                  __bf16* __restrict__ k, __bf16* __restrict__ v)
{
    size_t i = (size_t)blockIdx.x * 256 + threadIdx.x;   // over B*T*E
    int c = (int)(i % E_);
    size_t bt = i / E_;
    int t = (int)(bt % T_);
    int b = (int)(bt / T_);
    int hh = c / DH_, d = c % DH_;
    size_t dst  = (((size_t)b * H_ + hh) * T_ + t) * DH_ + d;
    size_t base = bt * E3_;
    q[dst] = qkv[base + c];
    k[dst] = qkv[base + E_ + c];
    v[dst] = qkv[base + 2 * E_ + c];
}

// ---------------- flash attention ----------------
// grid (T/128, B*H), 256 threads = 8 waves, each wave owns 16 query rows.
__global__ __launch_bounds__(256)
void attn_kernel(const __bf16* __restrict__ Q, const __bf16* __restrict__ Kmat,
                 const __bf16* __restrict__ Vmat, __bf16* __restrict__ Y)
{
    const int LD = 80;                               // bf16 stride, 160B (16B aligned)
    __shared__ __bf16 Qs[128 * 80];                  // [q][dh]
    __shared__ __bf16 Ks[64 * 80];                   // [key][dh]
    __shared__ __bf16 Vt[64 * 80];                   // [dh][key]  (transposed)
    __shared__ __bf16 Ps[128 * 80];                  // per-wave 16 rows: [q][key]

    int tid = threadIdx.x, lane = tid & 31, wave = tid >> 5;
    int q0 = blockIdx.x * 128;
    int bh = blockIdx.y;
    int b = bh >> 3, h = bh & 7;
    const __bf16* qb = Q    + (size_t)bh * T_ * DH_;
    const __bf16* kb = Kmat + (size_t)bh * T_ * DH_;
    const __bf16* vb = Vmat + (size_t)bh * T_ * DH_;

    // stage Q tile 128x64 async (1024 x 16B chunks)
#pragma unroll
    for (int j = 0; j < 4; j++) {
        int u = tid + 256 * j;
        int row = u >> 3, ch = u & 7;
        async_copy16(&Qs[row * LD + ch * 8],
                     qb + (size_t)(q0 + row) * DH_ + ch * 8);
    }

    int mlane = lane & 15;
    int ah  = (lane >> 4) * 8;
    int bhh = (lane >> 4) * 8;
    int mh  = (lane >> 4) * 8;
    int wq  = wave * 16;

    v8f o[4];
#pragma unroll
    for (int dt = 0; dt < 4; dt++) o[dt] = {};
    float mrow[8], lrow[8];
#pragma unroll
    for (int e = 0; e < 8; e++) { mrow[e] = -1e30f; lrow[e] = 0.0f; }

    int nkt = (q0 >> 6) + 2;                         // causal: key tiles of 64
    for (int kt = 0; kt < nkt; kt++) {
        __syncthreads();
        // stage K tile 64x64 async (512 x 16B chunks)
#pragma unroll
        for (int j = 0; j < 2; j++) {
            int u = tid + 256 * j;
            int row = u >> 3, ch = u & 7;
            async_copy16(&Ks[row * LD + ch * 8],
                         kb + (size_t)(kt * 64 + row) * DH_ + ch * 8);
        }
        // stage V tile transposed: Vt[dh][key] (b128 load + b16 scatter)
#pragma unroll
        for (int j = 0; j < 2; j++) {
            int u = tid + 256 * j;
            int key = u >> 3, ch = u & 7;
            U4H w;
            w.u = *(const uint4*)(vb + (size_t)(kt * 64 + key) * DH_ + ch * 8);
#pragma unroll
            for (int i = 0; i < 8; i++)
                Vt[(ch * 8 + i) * LD + key] = w.h[i];
        }
        async_wait0();
        __syncthreads();

        // S = Q K^T  (wave: 16 q-rows x 64 keys, 4 n-tiles, 2 k-steps over DH)
        v8f s[4];
#pragma unroll
        for (int nt = 0; nt < 4; nt++) s[nt] = {};
        for (int ks = 0; ks < 64; ks += 32) {
            v16bf a = ld16(&Qs[(wq + mlane) * LD + ks + ah],
                           &Qs[(wq + mlane) * LD + ks + ah + 16]);
#pragma unroll
            for (int nt = 0; nt < 4; nt++) {
                v16bf bo = ld16(&Ks[(nt * 16 + mlane) * LD + ks + bhh],
                                &Ks[(nt * 16 + mlane) * LD + ks + bhh + 16]);
                s[nt] = __builtin_amdgcn_wmma_f32_16x16x32_bf16(
                    false, a, false, bo, (short)0, s[nt], false, false);
            }
        }

        // scale, causal mask, online softmax (row stats per (e, half))
#pragma unroll
        for (int e = 0; e < 8; e++) {
            int qrow = q0 + wq + mh + e;
            float mx = -1e30f;
#pragma unroll
            for (int nt = 0; nt < 4; nt++) {
                int kcol = kt * 64 + nt * 16 + mlane;
                float sv = s[nt][e] * 0.125f;        // 1/sqrt(64)
                if (kcol > qrow) sv = -1e30f;
                s[nt][e] = sv;
                mx = fmaxf(mx, sv);
            }
#pragma unroll
            for (int off = 1; off < 16; off <<= 1) mx = fmaxf(mx, __shfl_xor(mx, off, 16));
            float mnew  = fmaxf(mrow[e], mx);
            float scale = __expf(mrow[e] - mnew);
            float rs = 0.0f;
#pragma unroll
            for (int nt = 0; nt < 4; nt++) {
                float p = __expf(s[nt][e] - mnew);
                s[nt][e] = p;
                rs += p;
            }
#pragma unroll
            for (int off = 1; off < 16; off <<= 1) rs += __shfl_xor(rs, off, 16);
            lrow[e] = lrow[e] * scale + rs;
            mrow[e] = mnew;
#pragma unroll
            for (int dt = 0; dt < 4; dt++) o[dt][e] *= scale;
        }

        // C-layout -> A-layout via per-wave LDS region
#pragma unroll
        for (int nt = 0; nt < 4; nt++)
#pragma unroll
            for (int e = 0; e < 8; e++)
                Ps[(wq + mh + e) * LD + nt * 16 + mlane] = (__bf16)s[nt][e];

        // O += P @ V  (4 dh-tiles, 2 k-steps over 64 keys)
        for (int ks = 0; ks < 64; ks += 32) {
            v16bf a = ld16(&Ps[(wq + mlane) * LD + ks + ah],
                           &Ps[(wq + mlane) * LD + ks + ah + 16]);
#pragma unroll
            for (int dt = 0; dt < 4; dt++) {
                v16bf bo = ld16(&Vt[(dt * 16 + mlane) * LD + ks + bhh],
                                &Vt[(dt * 16 + mlane) * LD + ks + bhh + 16]);
                o[dt] = __builtin_amdgcn_wmma_f32_16x16x32_bf16(
                    false, a, false, bo, (short)0, o[dt], false, false);
            }
        }
    }

    // normalize and scatter (bf16) to y [B,T,E] at head slice
#pragma unroll
    for (int e = 0; e < 8; e++) {
        float inv = 1.0f / lrow[e];
        size_t row = (size_t)b * T_ + (q0 + wq + mh + e);
#pragma unroll
        for (int dt = 0; dt < 4; dt++)
            Y[row * E_ + h * DH_ + dt * 16 + mlane] = (__bf16)(o[dt][e] * inv);
    }
}

// ---------------- logits GEMV: out[b,v] = dot(hlast[b], wte[v]) ----------------
__global__ __launch_bounds__(256)
void logits_kernel(const float* __restrict__ hlast, const float* __restrict__ wte,
                   float* __restrict__ out)
{
    int gw   = (int)((blockIdx.x * 256 + threadIdx.x) >> 5);
    int lane = threadIdx.x & 31;
    if (gw >= B_ * V_) return;
    int b = gw / V_, v = gw % V_;
    const float* wr = wte + (size_t)v * E_;
    const float* hr = hlast + (size_t)b * E_;
    float s = 0.0f;
#pragma unroll
    for (int j = 0; j < 16; j++) s += wr[lane + 32*j] * hr[lane + 32*j];
#pragma unroll
    for (int off = 16; off >= 1; off >>= 1) s += __shfl_xor(s, off, 32);
    if (lane == 0) out[gw] = s;
}

// ---------------- launch ----------------
extern "C" void kernel_launch(void* const* d_in, const int* in_sizes, int n_in,
                              void* d_out, int out_size, void* d_ws, size_t ws_size,
                              hipStream_t stream)
{
    const int*   idx   = (const int*)  d_in[0];
    const float* wte   = (const float*)d_in[1];
    const float* wpe   = (const float*)d_in[2];
    const float* ln1_g = (const float*)d_in[3];
    const float* ln1_b = (const float*)d_in[4];
    const float* Wqkv  = (const float*)d_in[5];
    const float* bqkv  = (const float*)d_in[6];
    const float* Wo    = (const float*)d_in[7];
    const float* bo    = (const float*)d_in[8];
    const float* ln2_g = (const float*)d_in[9];
    const float* ln2_b = (const float*)d_in[10];
    const float* Wfc   = (const float*)d_in[11];
    const float* bfc   = (const float*)d_in[12];
    const float* Wfc2  = (const float*)d_in[13];
    const float* bfc2  = (const float*)d_in[14];
    const float* lnf_g = (const float*)d_in[15];
    const float* lnf_b = (const float*)d_in[16];

    char* ws = (char*)d_ws;
    size_t off = 0;
    auto alloc = [&](size_t bytes) -> void* {
        void* p = ws + off;
        off += (bytes + 255) & ~(size_t)255;
        return p;
    };
    const size_t MT = (size_t)B_ * T_;
    float*  x      = (float*) alloc(MT * E_  * 4);
    __bf16* hbuf   = (__bf16*)alloc(MT * E_  * 2);
    __bf16* qkv    = (__bf16*)alloc(MT * E3_ * 2);
    __bf16* qb     = (__bf16*)alloc(MT * E_  * 2);
    __bf16* kb     = (__bf16*)alloc(MT * E_  * 2);
    __bf16* vb     = (__bf16*)alloc(MT * E_  * 2);
    __bf16* y      = (__bf16*)alloc(MT * E_  * 2);
    __bf16* mid    = (__bf16*)alloc(MT * E4_ * 2);
    float*  hlast  = (float*) alloc((size_t)B_ * E_ * 4);
    __bf16* Wqkv_b = (__bf16*)alloc((size_t)L_ * E_  * E3_ * 2);
    __bf16* Wo_b   = (__bf16*)alloc((size_t)L_ * E_  * E_  * 2);
    __bf16* Wfc_b  = (__bf16*)alloc((size_t)L_ * E_  * E4_ * 2);
    __bf16* Wfc2_b = (__bf16*)alloc((size_t)L_ * E4_ * E_  * 2);

    const int M = B_ * T_;                        // 4096

    // one-shot weight conversion (L2-resident afterwards)
    {
        int n1 = L_ * E_ * E3_;
        int n2 = L_ * E_ * E_;
        int n3 = L_ * E_ * E4_;
        convert_kernel<<<(n1 + 255) / 256, 256, 0, stream>>>(Wqkv, Wqkv_b, n1);
        convert_kernel<<<(n2 + 255) / 256, 256, 0, stream>>>(Wo,   Wo_b,   n2);
        convert_kernel<<<(n3 + 255) / 256, 256, 0, stream>>>(Wfc,  Wfc_b,  n3);
        convert_kernel<<<(n3 + 255) / 256, 256, 0, stream>>>(Wfc2, Wfc2_b, n3);
    }

    embed_kernel<<<(int)(MT * E_ / 256), 256, 0, stream>>>(idx, wte, wpe, x);

    for (int l = 0; l < L_; l++) {
        ln_kernel<<<M / 8, 256, 0, stream>>>(x, ln1_g + l * E_, ln1_b + l * E_, hbuf, M);
        gemm_kernel<false, false, true><<<dim3(E3_ / 128, M / 128), 256, 0, stream>>>(
            hbuf, Wqkv_b + (size_t)l * E_ * E3_, bqkv + (size_t)l * E3_,
            nullptr, qkv, M, E3_, E_);
        split_kernel<<<(int)(MT * E_ / 256), 256, 0, stream>>>(qkv, qb, kb, vb);
        attn_kernel<<<dim3(T_ / 128, B_ * H_), 256, 0, stream>>>(qb, kb, vb, y);
        gemm_kernel<false, true, false><<<dim3(E_ / 128, M / 128), 256, 0, stream>>>(
            y, Wo_b + (size_t)l * E_ * E_, bo + (size_t)l * E_,
            x, x, M, E_, E_);
        ln_kernel<<<M / 8, 256, 0, stream>>>(x, ln2_g + l * E_, ln2_b + l * E_, hbuf, M);
        gemm_kernel<true, false, true><<<dim3(E4_ / 128, M / 128), 256, 0, stream>>>(
            hbuf, Wfc_b + (size_t)l * E_ * E4_, bfc + (size_t)l * E4_,
            nullptr, mid, M, E4_, E_);
        gemm_kernel<false, true, false><<<dim3(E_ / 128, M / 128), 256, 0, stream>>>(
            mid, Wfc2_b + (size_t)l * E4_ * E_, bfc2 + (size_t)l * E_,
            x, x, M, E_, E4_);
    }

    lnf_last_kernel<<<1, 64, 0, stream>>>(x, lnf_g, lnf_b, hlast);
    logits_kernel<<<B_ * V_ / 8, 256, 0, stream>>>(hlast, wte, (float*)d_out);
}
